// GQAAttention_34935263986350
// MI455X (gfx1250) — matrix operand order repaired and checked by reading, not compiled, not run on hardware
//
#include <hip/hip_runtime.h>

// GQA attention for MI455X (gfx1250, wave32).
// ~155 GFLOP vs ~115 MB unique traffic -> matrix-op bound; all matrix math on
// v_wmma_f32_16x16x32_f16. Operands staged in LDS (320KB/WGP) with
// double-buffered global_load_async_to_lds_b128 (ASYNCcnt) so DMA of the next
// slab overlaps WMMA on the current one; WMMA fragments fed by ds_load_b128.

typedef __attribute__((ext_vector_type(16))) _Float16 v16h;
typedef __attribute__((ext_vector_type(8)))  _Float16 v8h;
typedef __attribute__((ext_vector_type(4)))  _Float16 v4h;
typedef __attribute__((ext_vector_type(8)))  float    v8f;

constexpr int kB   = 2;
constexpr int kS   = 2048;
constexpr int kHid = 2048;
constexpr int kNH  = 32;
constexpr int kNKV = 8;
constexpr int kHD  = 64;
constexpr int kLdsPitch = 72;  // 64 f16 + 8 pad -> 144B rows, kills bank conflicts

// ---------------- WMMA fragment loaders (CDNA5 16-bit layouts, ISA 7.12.2) --
// A (16x32, M=lane%16): VGPR0-3 <- K = (lane>>4)*8 + 0..7 ; VGPR4-7 <- +16.
static __device__ __forceinline__ v16h load_frag_a(const _Float16* base, int ldk) {
  const int lane = threadIdx.x & 31;
  const _Float16* p = base + (size_t)(lane & 15) * ldk + ((lane >> 4) * 8);
  v16h r;
  *((v8h*)&r)     = *(const v8h*)(p);
  *((v8h*)&r + 1) = *(const v8h*)(p + 16);
  return r;
}
// B (32x16, N=lane%16): VGPR0-3 <- K = (lane>>4)*8 + 0..7 ; VGPR4-7 <- +16.
static __device__ __forceinline__ v16h load_frag_b(const _Float16* base, int ldk) {
  const int lane = threadIdx.x & 31;
  const _Float16* p = base + (size_t)(lane & 15) * ldk + ((lane >> 4) * 8);
  v16h r;
  *((v8h*)&r)     = *(const v8h*)(p);
  *((v8h*)&r + 1) = *(const v8h*)(p + 16);
  return r;
}

static __device__ __forceinline__ v8f wmma_f16(v16h a, v16h b, v8f c) {
  return __builtin_amdgcn_wmma_f32_16x16x32_f16(false, a, false, b, (short)0, c,
                                                false, false);
}

// Async DMA of a [rows x 64] f16 tile (row stride ldk elems in global) into
// LDS with 72-f16 padded rows. 16B per lane per instruction; ASYNCcnt-tracked.
static __device__ __forceinline__ void async_tile_64(
    const _Float16* __restrict__ g, int ldk, uint32_t lds_base, int rows,
    int tid, int nthreads) {
  const int total = rows * 8;  // 16B chunks
  for (int c = tid; c < total; c += nthreads) {
    const int row = c >> 3, ch = c & 7;
    const _Float16* gp = g + (size_t)row * ldk + ch * 8;
    const uint32_t la = lds_base + (uint32_t)(row * (kLdsPitch * 2) + ch * 16);
    asm volatile("global_load_async_to_lds_b128 %0, %1, off"
                 :: "v"(la), "v"(gp) : "memory");
  }
}
static __device__ __forceinline__ void wait_async0() {
  asm volatile("s_wait_asynccnt 0" ::: "memory");
}

// ---------------- f32 -> f16 convert (x4 vectorized) ------------------------
__global__ void cvt_f32_f16_kernel(const float* __restrict__ in,
                                   _Float16* __restrict__ out, int n4) {
  int i = blockIdx.x * blockDim.x + threadIdx.x;
  if (i >= n4) return;
  float4 f = ((const float4*)in)[i];
  v4h o;
  o.x = (_Float16)f.x; o.y = (_Float16)f.y;
  o.z = (_Float16)f.z; o.w = (_Float16)f.w;
  ((v4h*)out)[i] = o;
}

// ---------------- LDS-staged WMMA GEMM: C[m,n] = sum_k A[m,k]*Bw[n,k] -------
// 128x128 tile per 8-wave block; each wave owns a 64x32 sub-tile.
// K staged in 64-wide slabs, double-buffered via async loads.
// MODE 0: f32 row-major out (O-proj -> d_out)
// MODE 2: f16 out in [b, head, s, d]   (Q, K)
// MODE 3: f16 out in [b, head, d, s]   (V transposed: contiguous P@V loads)
template <int MODE>
__global__ __launch_bounds__(256) void gemm_wmma_kernel(
    const _Float16* __restrict__ A, const _Float16* __restrict__ Bw,
    void* __restrict__ out, int M, int N, int K, int nheads) {
  __shared__ __align__(16) _Float16 sA[2][128 * kLdsPitch];
  __shared__ __align__(16) _Float16 sB[2][128 * kLdsPitch];
  const int tid  = threadIdx.x;
  const int lane = tid & 31;
  const int wave = tid >> 5;
  const int mw   = wave >> 2;  // 0..1 -> 64-row half
  const int nw   = wave & 3;   // 0..3 -> 32-col quarter
  const int nb   = N >> 7;
  const int tm   = (blockIdx.x / nb) << 7;
  const int tn   = (blockIdx.x % nb) << 7;

  const uint32_t lA[2] = {(uint32_t)(size_t)&sA[0][0], (uint32_t)(size_t)&sA[1][0]};
  const uint32_t lB[2] = {(uint32_t)(size_t)&sB[0][0], (uint32_t)(size_t)&sB[1][0]};

  auto stage = [&](int it, int buf) {
    async_tile_64(A + (size_t)tm * K + it * 64, K, lA[buf], 128, tid, 256);
    async_tile_64(Bw + (size_t)tn * K + it * 64, K, lB[buf], 128, tid, 256);
  };

  v8f acc[4][2] = {};
  const int nk = K >> 6;
  stage(0, 0);
  for (int it = 0; it < nk; ++it) {
    const int cur = it & 1;
    wait_async0();
    __syncthreads();  // buf[cur] ready; everyone done reading buf[cur^1]
    if (it + 1 < nk) stage(it + 1, cur ^ 1);
    const _Float16* Ab = &sA[cur][0] + (size_t)(mw * 64) * kLdsPitch;
    const _Float16* Bb = &sB[cur][0] + (size_t)(nw * 32) * kLdsPitch;
#pragma unroll
    for (int kk = 0; kk < 64; kk += 32) {
      v16h a[4], b[2];
#pragma unroll
      for (int mi = 0; mi < 4; ++mi)
        a[mi] = load_frag_a(Ab + (size_t)(mi * 16) * kLdsPitch + kk, kLdsPitch);
#pragma unroll
      for (int ni = 0; ni < 2; ++ni)
        b[ni] = load_frag_b(Bb + (size_t)(ni * 16) * kLdsPitch + kk, kLdsPitch);
#pragma unroll
      for (int ni = 0; ni < 2; ++ni)
#pragma unroll
        for (int mi = 0; mi < 4; ++mi)
          acc[mi][ni] = wmma_f16(a[mi], b[ni], acc[mi][ni]);
    }
  }

  const int lh = lane >> 4;
#pragma unroll
  for (int mi = 0; mi < 4; ++mi)
#pragma unroll
    for (int ni = 0; ni < 2; ++ni)
#pragma unroll
      for (int r = 0; r < 8; ++r) {
        int row = tm + mw * 64 + mi * 16 + r + 8 * lh;  // m (= b*S + s)
        int col = tn + nw * 32 + ni * 16 + (lane & 15); // n
        float v = acc[mi][ni][r];
        if (MODE == 0) {
          ((float*)out)[(size_t)row * N + col] = v;
        } else {
          int bb = row >> 11, s = row & (kS - 1);
          int h  = col >> 6,  d = col & (kHD - 1);
          size_t idx;
          if (MODE == 2)
            idx = (((size_t)bb * nheads + h) * kS + s) * kHD + d;
          else
            idx = (((size_t)bb * nheads + h) * kHD + d) * kS + s;
          ((_Float16*)out)[idx] = (_Float16)v;
        }
      }
}

// ---------------- RoPE in-place on f16 [b,h,s,d] ----------------------------
__global__ void rope_kernel(_Float16* __restrict__ q,
                            const float* __restrict__ cosb,
                            const float* __restrict__ sinb, int total) {
  int i = blockIdx.x * blockDim.x + threadIdx.x;
  if (i >= total) return;
  int d  = i & 31;
  int s  = (i >> 5) & (kS - 1);
  int bh = i >> 16;  // 2048*32 = 2^16 elems per (b,h)
  size_t base = ((size_t)bh * kS + s) * kHD;
  float x1 = (float)q[base + d];
  float x2 = (float)q[base + d + 32];
  float c1 = cosb[s * kHD + d],      s1 = sinb[s * kHD + d];
  float c2 = cosb[s * kHD + d + 32], s2 = sinb[s * kHD + d + 32];
  q[base + d]      = (_Float16)(x1 * c1 - x2 * s1);
  q[base + d + 32] = (_Float16)(x2 * c2 + x1 * s2);
}

// ---------------- flash attention -------------------------------------------
// Block = (b, h, 128 q rows); 4 waves x 32 q rows. K/V 64-tiles staged in LDS
// via async DMA (shared by all 4 waves), double-buffered across the kv loop.
// Each K/V fragment read from LDS feeds 2 WMMAs (one per q sub-tile).
__global__ __launch_bounds__(128) void attn_kernel(
    const _Float16* __restrict__ Qt,   // [b, nh, s, d]
    const _Float16* __restrict__ Kt,   // [b, nkv, s, d]
    const _Float16* __restrict__ Vt,   // [b, nkv, d, s]
    _Float16* __restrict__ AO) {       // [b, s, nh*hd]
  __shared__ __align__(16) _Float16 sK[2][64 * kLdsPitch];
  __shared__ __align__(16) _Float16 sV[2][64 * kLdsPitch];
  __shared__ __align__(16) _Float16 pbuf[4][32 * 64];
  const int tid  = threadIdx.x;
  const int lane = tid & 31;
  const int wave = tid >> 5;
  const int qb   = blockIdx.x & 15;          // S/128 = 16 q blocks
  const int h    = (blockIdx.x >> 4) & 31;
  const int b    = blockIdx.x >> 9;
  const int hk   = h >> 2;                   // N_REP = 4

  const _Float16* Qh = Qt + ((size_t)(b * kNH + h) * kS + qb * 128 + wave * 32) * kHD;
  const _Float16* Kh = Kt + (size_t)(b * kNKV + hk) * kS * kHD;
  const _Float16* Vh = Vt + (size_t)(b * kNKV + hk) * kHD * kS;

  const uint32_t lK[2] = {(uint32_t)(size_t)&sK[0][0], (uint32_t)(size_t)&sK[1][0]};
  const uint32_t lV[2] = {(uint32_t)(size_t)&sV[0][0], (uint32_t)(size_t)&sV[1][0]};
  auto stage = [&](int it, int buf) {
    async_tile_64(Kh + (size_t)(it * 64) * kHD, kHD, lK[buf], 64, tid, 128);
    async_tile_64(Vh + it * 64, kS, lV[buf], 64, tid, 128);
  };

  v16h qa[2][2];
#pragma unroll
  for (int mt = 0; mt < 2; ++mt) {
    qa[mt][0] = load_frag_a(Qh + (size_t)(mt * 16) * kHD + 0, kHD);
    qa[mt][1] = load_frag_a(Qh + (size_t)(mt * 16) * kHD + 32, kHD);
  }

  const int lh = lane >> 4;
  float mrow[2][8], lrow[2][8];
  v8f oacc[2][4] = {};
#pragma unroll
  for (int mt = 0; mt < 2; ++mt)
#pragma unroll
    for (int r = 0; r < 8; ++r) { mrow[mt][r] = -1e30f; lrow[mt][r] = 0.f; }
  _Float16* pb = &pbuf[wave][0];

  stage(0, 0);
  for (int it = 0; it < kS / 64; ++it) {
    const int cur = it & 1;
    wait_async0();
    __syncthreads();
    if (it + 1 < kS / 64) stage(it + 1, cur ^ 1);
    const _Float16* Kb = &sK[cur][0];
    const _Float16* Vb = &sV[cur][0];

    // scores 32x64 = Q @ K^T; each K fragment feeds both q sub-tiles
    v8f sc[2][4] = {};
#pragma unroll
    for (int ni = 0; ni < 4; ++ni) {
      v16h bk0 = load_frag_b(Kb + (size_t)(ni * 16) * kLdsPitch + 0, kLdsPitch);
      v16h bk1 = load_frag_b(Kb + (size_t)(ni * 16) * kLdsPitch + 32, kLdsPitch);
#pragma unroll
      for (int mt = 0; mt < 2; ++mt) {
        sc[mt][ni] = wmma_f16(qa[mt][0], bk0, sc[mt][ni]);
        sc[mt][ni] = wmma_f16(qa[mt][1], bk1, sc[mt][ni]);
      }
    }
    // online softmax: row r(+8*lh) lives across the 16 lanes of a half-wave
#pragma unroll
    for (int mt = 0; mt < 2; ++mt)
#pragma unroll
      for (int r = 0; r < 8; ++r) {
        float s0 = sc[mt][0][r] * 0.125f, s1 = sc[mt][1][r] * 0.125f;
        float s2 = sc[mt][2][r] * 0.125f, s3 = sc[mt][3][r] * 0.125f;
        float tmax = fmaxf(fmaxf(s0, s1), fmaxf(s2, s3));
        tmax = fmaxf(tmax, __shfl_xor(tmax, 1, 32));
        tmax = fmaxf(tmax, __shfl_xor(tmax, 2, 32));
        tmax = fmaxf(tmax, __shfl_xor(tmax, 4, 32));
        tmax = fmaxf(tmax, __shfl_xor(tmax, 8, 32));
        float mnew = fmaxf(mrow[mt][r], tmax);
        float corr = __expf(mrow[mt][r] - mnew);
        mrow[mt][r] = mnew;
        float p0 = __expf(s0 - mnew), p1 = __expf(s1 - mnew);
        float p2 = __expf(s2 - mnew), p3 = __expf(s3 - mnew);
        sc[mt][0][r] = p0; sc[mt][1][r] = p1;
        sc[mt][2][r] = p2; sc[mt][3][r] = p3;
        float psum = p0 + p1 + p2 + p3;
        psum += __shfl_xor(psum, 1, 32);
        psum += __shfl_xor(psum, 2, 32);
        psum += __shfl_xor(psum, 4, 32);
        psum += __shfl_xor(psum, 8, 32);
        lrow[mt][r] = lrow[mt][r] * corr + psum;
#pragma unroll
        for (int dt = 0; dt < 4; ++dt) oacc[mt][dt][r] *= corr;
      }
    // P: C-layout -> A-layout via per-wave LDS tile (in-order LDS, same wave)
#pragma unroll
    for (int mt = 0; mt < 2; ++mt)
#pragma unroll
      for (int ni = 0; ni < 4; ++ni)
#pragma unroll
        for (int r = 0; r < 8; ++r)
          pb[(mt * 16 + r + 8 * lh) * 64 + ni * 16 + (lane & 15)] =
              (_Float16)sc[mt][ni][r];
    asm volatile("s_wait_dscnt 0" ::: "memory");
    v16h pa[2][2];
#pragma unroll
    for (int mt = 0; mt < 2; ++mt) {
      pa[mt][0] = load_frag_a(pb + (size_t)(mt * 16) * 64 + 0, 64);
      pa[mt][1] = load_frag_a(pb + (size_t)(mt * 16) * 64 + 32, 64);
    }
    // out += P(32x64) @ V(64x64); each V fragment feeds both q sub-tiles
#pragma unroll
    for (int dt = 0; dt < 4; ++dt) {
      v16h bv0 = load_frag_b(Vb + (size_t)(dt * 16) * kLdsPitch + 0, kLdsPitch);
      v16h bv1 = load_frag_b(Vb + (size_t)(dt * 16) * kLdsPitch + 32, kLdsPitch);
#pragma unroll
      for (int mt = 0; mt < 2; ++mt) {
        oacc[mt][dt] = wmma_f16(pa[mt][0], bv0, oacc[mt][dt]);
        oacc[mt][dt] = wmma_f16(pa[mt][1], bv1, oacc[mt][dt]);
      }
    }
  }
  // normalize + store f16 attention output in [b, s, h*64+d]
#pragma unroll
  for (int mt = 0; mt < 2; ++mt)
#pragma unroll
    for (int dt = 0; dt < 4; ++dt)
#pragma unroll
      for (int r = 0; r < 8; ++r) {
        int s   = qb * 128 + wave * 32 + mt * 16 + r + 8 * lh;
        int col = h * kHD + dt * 16 + (lane & 15);
        AO[((size_t)b * kS + s) * (kNH * kHD) + col] =
            (_Float16)(oacc[mt][dt][r] / lrow[mt][r]);
      }
}

// ---------------- host side -------------------------------------------------
extern "C" void kernel_launch(void* const* d_in, const int* in_sizes, int n_in,
                              void* d_out, int out_size, void* d_ws,
                              size_t ws_size, hipStream_t stream) {
  const float* hs   = (const float*)d_in[0];
  const float* cosb = (const float*)d_in[1];
  const float* sinb = (const float*)d_in[2];
  const float* wq   = (const float*)d_in[3];
  const float* wk   = (const float*)d_in[4];
  const float* wv   = (const float*)d_in[5];
  const float* wo   = (const float*)d_in[6];

  char* ws = (char*)d_ws;
  size_t off = 0;
  auto alloc = [&](size_t bytes) -> void* {
    void* p = ws + off;
    off += (bytes + 255) & ~(size_t)255;
    return p;
  };
  const size_t M = (size_t)kB * kS;  // 4096
  _Float16* Xh  = (_Float16*)alloc(M * kHid * 2);
  _Float16* Wqh = (_Float16*)alloc((size_t)kHid * kHid * 2);
  _Float16* Wkh = (_Float16*)alloc((size_t)kNKV * kHD * kHid * 2);
  _Float16* Wvh = (_Float16*)alloc((size_t)kNKV * kHD * kHid * 2);
  _Float16* Woh = (_Float16*)alloc((size_t)kHid * kHid * 2);
  _Float16* Qt  = (_Float16*)alloc((size_t)kB * kNH * kS * kHD * 2);
  _Float16* Kt  = (_Float16*)alloc((size_t)kB * kNKV * kS * kHD * 2);
  _Float16* Vt  = (_Float16*)alloc((size_t)kB * kNKV * kS * kHD * 2);
  _Float16* AO  = (_Float16*)alloc(M * kHid * 2);
  (void)ws_size;  // ~80 MB used

  auto cvt = [&](const float* src, _Float16* dst, size_t n) {
    int n4 = (int)(n / 4);
    cvt_f32_f16_kernel<<<(n4 + 255) / 256, 256, 0, stream>>>(src, dst, n4);
  };
  cvt(hs, Xh, M * kHid);
  cvt(wq, Wqh, (size_t)kHid * kHid);
  cvt(wk, Wkh, (size_t)kNKV * kHD * kHid);
  cvt(wv, Wvh, (size_t)kNKV * kHD * kHid);
  cvt(wo, Woh, (size_t)kHid * kHid);

  // QKV projections: 128x128 tile per 256-thread block
  gemm_wmma_kernel<2><<<(int)(M / 128) * (kHid / 128), 256, 0, stream>>>(
      Xh, Wqh, Qt, (int)M, kHid, kHid, kNH);
  gemm_wmma_kernel<2><<<(int)(M / 128) * ((kNKV * kHD) / 128), 256, 0, stream>>>(
      Xh, Wkh, Kt, (int)M, kNKV * kHD, kHid, kNKV);
  gemm_wmma_kernel<3><<<(int)(M / 128) * ((kNKV * kHD) / 128), 256, 0, stream>>>(
      Xh, Wvh, Vt, (int)M, kNKV * kHD, kHid, kNKV);

  // RoPE (in-place, f32 math)
  {
    int tq = kB * kNH * kS * 32;
    rope_kernel<<<(tq + 255) / 256, 256, 0, stream>>>(Qt, cosb, sinb, tq);
    int tk = kB * kNKV * kS * 32;
    rope_kernel<<<(tk + 255) / 256, 256, 0, stream>>>(Kt, cosb, sinb, tk);
  }
  // flash attention: block = (b, h, q128) -> 1024 blocks of 4 waves
  attn_kernel<<<kB * kNH * (kS / 128), 128, 0, stream>>>(Qt, Kt, Vt, AO);
  // output projection -> f32 d_out
  gemm_wmma_kernel<0><<<(int)(M / 128) * (kHid / 128), 256, 0, stream>>>(
      AO, Woh, d_out, (int)M, kHid, kHid, 0);
}
